// RobertaSelfAttention_matchKV_34703335751974
// MI455X (gfx1250) — matrix operand order repaired and checked by reading, not compiled, not run on hardware
//
#include <hip/hip_runtime.h>

// ---------------------------------------------------------------------------
// RobertaSelfAttention_matchKV for MI455X (gfx1250, wave32, WMMA)
//
// Pipeline:
//   1) wconv_kernel      : Wk, Wv f32 -> bf16 (read once, reused by 256 M-blocks)
//   2) kvproj_kernel     : dual bf16 WMMA GEMM (K-proj and V-proj share the H
//                          tile in LDS). Emits v_wmma_f32_16x16x32_bf16.
//                          V1 = relu(H@Wv^T+bv) stored f32; dot(relu(K),rh)
//                          reduced in-register (shfl_xor) + LDS, stored f32.
//   3) scan_gather_kernel: blocked associative prefix over the "last two valid
//                          indices" operator, then weighted double-gather from
//                          V1 (L2-resident, 192MB L2 >> 100MB V1).
// ---------------------------------------------------------------------------

typedef __bf16 bf16_t;
typedef __attribute__((ext_vector_type(16))) __bf16 v16bf;
typedef __attribute__((ext_vector_type(8)))  __bf16 v8bf;
typedef __attribute__((ext_vector_type(8)))  float  v8f;

#define N_HEADS   12
#define HEAD_DIM  64
#define HIDDEN    768
#define BSZ       8
#define LEN       4096
#define M_TOTAL   (BSZ*LEN)      /* 32768 */
#define BM        128
#define BK        32

struct __align__(8) bf16x4 { bf16_t x, y, z, w; };

static __device__ __forceinline__ v16bf cat16(v8bf lo, v8bf hi) {
  v16bf r;
#pragma unroll
  for (int i = 0; i < 8; ++i) { r[i] = lo[i]; r[i + 8] = hi[i]; }
  return r;
}

// ---------------------------------------------------------------------------
// Kernel 1: weights f32 -> bf16
// ---------------------------------------------------------------------------
__global__ __launch_bounds__(256) void wconv_kernel(
    const float* __restrict__ Wk, const float* __restrict__ Wv,
    bf16_t* __restrict__ WkB, bf16_t* __restrict__ WvB) {
  int i = (blockIdx.x * 256 + threadIdx.x) * 4;
  float4 a = *(const float4*)(Wk + i);
  float4 b = *(const float4*)(Wv + i);
  bf16x4 ab = {(bf16_t)a.x, (bf16_t)a.y, (bf16_t)a.z, (bf16_t)a.w};
  bf16x4 bb = {(bf16_t)b.x, (bf16_t)b.y, (bf16_t)b.z, (bf16_t)b.w};
  *(bf16x4*)(WkB + i) = ab;
  *(bf16x4*)(WvB + i) = bb;
}

// ---------------------------------------------------------------------------
// Kernel 2: fused dual GEMM (K-proj + V-proj) with bf16 WMMA.
// Grid: (M_TOTAL/BM, N_HEADS). One block owns 128 rows x one full head (64 n).
// 8 waves, each wave a 32x32 region (2x2 WMMA tiles) of both outputs.
// ---------------------------------------------------------------------------
__global__ __launch_bounds__(256) void kvproj_kernel(
    const float*  __restrict__ H,
    const bf16_t* __restrict__ WkB, const bf16_t* __restrict__ WvB,
    const float*  __restrict__ bk,  const float*  __restrict__ bv,
    const float*  __restrict__ RH,
    float* __restrict__ V1, float* __restrict__ dotg) {
  __shared__ __align__(16) bf16_t lA[BM * BK];         // 8 KB
  __shared__ __align__(16) bf16_t lWk[HEAD_DIM * BK];  // 4 KB
  __shared__ __align__(16) bf16_t lWv[HEAD_DIM * BK];  // 4 KB
  __shared__ float dotLds[BM];

  const int tid  = threadIdx.x;
  const int lane = tid & 31;
  const int lr   = lane & 15;   // row/col index inside 16-wide fragment
  const int half = lane >> 4;   // selects K sub-range / M sub-range
  const int wave = tid >> 5;
  const int wm   = (wave & 3) * 32;   // wave M offset inside block
  const int wn   = (wave >> 2) * 32;  // wave N offset inside block (0 or 32)
  const int mBlock = blockIdx.x * BM;
  const int head   = blockIdx.y;

  if (tid < BM) dotLds[tid] = 0.f;

  const v8f vzero = {0.f, 0.f, 0.f, 0.f, 0.f, 0.f, 0.f, 0.f};
  v8f accK[2][2], accV[2][2];
#pragma unroll
  for (int mt = 0; mt < 2; ++mt)
#pragma unroll
    for (int nt = 0; nt < 2; ++nt) { accK[mt][nt] = vzero; accV[mt][nt] = vzero; }

  for (int kk = 0; kk < HIDDEN; kk += BK) {
    __syncthreads();  // previous fragments consumed before LDS overwrite

    // ---- stage H tile: f32 global -> bf16 LDS (128 x 32) ----
#pragma unroll
    for (int i = 0; i < 4; ++i) {
      int c   = tid + i * 256;   // 0..1023 float4 chunks
      int row = c >> 3;          // 0..127
      int kq  = c & 7;           // float4 within 32-k
      float4 v = *(const float4*)&H[(size_t)(mBlock + row) * HIDDEN + kk + kq * 4];
      bf16x4 s = {(bf16_t)v.x, (bf16_t)v.y, (bf16_t)v.z, (bf16_t)v.w};
      *(bf16x4*)&lA[row * BK + kq * 4] = s;
    }
    // ---- stage Wk/Wv tiles: bf16 global -> bf16 LDS (64 x 32 each) ----
    {
      int row = tid >> 2;        // 0..63
      int kq  = tid & 3;         // 8-bf16 chunk within 32-k
      size_t g = (size_t)(head * HEAD_DIM + row) * HIDDEN + kk + kq * 8;
      *(v8bf*)&lWk[row * BK + kq * 8] = *(const v8bf*)&WkB[g];
      *(v8bf*)&lWv[row * BK + kq * 8] = *(const v8bf*)&WvB[g];
    }
    // prefetch next H tile into cache hierarchy (global_prefetch_b8)
    if (kk + BK < HIDDEN)
      __builtin_prefetch(&H[(size_t)(mBlock + (tid >> 1)) * HIDDEN + kk + BK], 0, 0);

    __syncthreads();

    // ---- load A fragments (contiguous k = 16*half + s, matches B) ----
    v16bf aF[2];
#pragma unroll
    for (int mt = 0; mt < 2; ++mt) {
      int ar = (wm + mt * 16 + lr) * BK + half * 16;
      aF[mt] = cat16(*(const v8bf*)&lA[ar], *(const v8bf*)&lA[ar + 8]);
    }
    // ---- load B fragments (lane = column n; half selects K range of 16) ----
    v16bf bkF[2], bvF[2];
#pragma unroll
    for (int nt = 0; nt < 2; ++nt) {
      int br = (wn + nt * 16 + lr) * BK + half * 16;
      bkF[nt] = cat16(*(const v8bf*)&lWk[br], *(const v8bf*)&lWk[br + 8]);
      bvF[nt] = cat16(*(const v8bf*)&lWv[br], *(const v8bf*)&lWv[br + 8]);
    }
    // ---- 8 WMMAs per k-step ----
#pragma unroll
    for (int mt = 0; mt < 2; ++mt)
#pragma unroll
      for (int nt = 0; nt < 2; ++nt) {
        accK[mt][nt] = __builtin_amdgcn_wmma_f32_16x16x32_bf16(
            false, aF[mt], false, bkF[nt], (short)0, accK[mt][nt], false, false);
        accV[mt][nt] = __builtin_amdgcn_wmma_f32_16x16x32_bf16(
            false, aF[mt], false, bvF[nt], (short)0, accV[mt][nt], false, false);
      }
  }

  // ---- epilogue: bias + relu, store V1, reduce dot(relu(K), ReadingHead) ----
  float bkv[2], bvv[2], rhv[2];
#pragma unroll
  for (int nt = 0; nt < 2; ++nt) {
    int nG  = head * HEAD_DIM + wn + nt * 16 + lr;
    bkv[nt] = bk[nG]; bvv[nt] = bv[nG]; rhv[nt] = RH[nG];
  }
#pragma unroll
  for (int mt = 0; mt < 2; ++mt) {
    float dsum[8];
#pragma unroll
    for (int j = 0; j < 8; ++j) dsum[j] = 0.f;
#pragma unroll
    for (int nt = 0; nt < 2; ++nt) {
      int nG = head * HEAD_DIM + wn + nt * 16 + lr;
#pragma unroll
      for (int j = 0; j < 8; ++j) {
        // C layout: VGPR j -> row M = j + 8*half, lane lr -> column
        float kv = fmaxf(accK[mt][nt][j] + bkv[nt], 0.f);
        float vv = fmaxf(accV[mt][nt][j] + bvv[nt], 0.f);
        int mG = mBlock + wm + mt * 16 + half * 8 + j;
        V1[(size_t)mG * HIDDEN + nG] = vv;
        dsum[j] += kv * rhv[nt];
      }
    }
    // butterfly-reduce across the 16 lanes that share a row (xor<=8 stays in half)
#pragma unroll
    for (int j = 0; j < 8; ++j) {
      float s = dsum[j];
      s += __shfl_xor(s, 1);
      s += __shfl_xor(s, 2);
      s += __shfl_xor(s, 4);
      s += __shfl_xor(s, 8);
      if (lr == 0)
        atomicAdd(&dotLds[wm + mt * 16 + half * 8 + j], s);  // ds_add_f32
    }
  }
  __syncthreads();
  if (tid < BM)
    dotg[(size_t)(mBlock + tid) * N_HEADS + head] = dotLds[tid];
}

// ---------------------------------------------------------------------------
// Kernel 3: blocked prefix scan over "last two valid indices" + gather.
// Grid: (N_HEADS, BSZ). One block per (b, head) sequence of 4096.
// ---------------------------------------------------------------------------
__global__ __launch_bounds__(256) void scan_gather_kernel(
    const float* __restrict__ dotg, const float* __restrict__ V1,
    const float* __restrict__ w1p,  const float* __restrict__ w2p,
    float* __restrict__ out) {
  __shared__ short sIdx0[LEN];
  __shared__ short sIdx1[LEN];
  __shared__ unsigned char sC[256];
  __shared__ short sJ0[256], sJ1[256];
  __shared__ short sI0[256], sI1[256];

  const int t    = threadIdx.x;
  const int head = blockIdx.x;
  const int b    = blockIdx.y;
  const int baseM = b * LEN;

  // phase 1: per-thread segment summary over 16 positions
  unsigned mask = 0;
  int c = 0, j0 = 0, j1 = 0;
#pragma unroll 4
  for (int i = 0; i < 16; ++i) {
    int l = t * 16 + i;
    float d = dotg[(size_t)(baseM + l) * N_HEADS + head];
    if (d > 0.5f) { mask |= 1u << i; j1 = j0; j0 = l; ++c; }
  }
  sC[t] = (unsigned char)c; sJ0[t] = (short)j0; sJ1[t] = (short)j1;
  __syncthreads();

  // phase 2: serial combine of 256 summaries (associative operator)
  if (t == 0) {
    int i0 = 0, i1 = 0;
    for (int s = 0; s < 256; ++s) {
      sI0[s] = (short)i0; sI1[s] = (short)i1;
      int cc = sC[s];
      if (cc == 1)      { i1 = i0; i0 = sJ0[s]; }
      else if (cc >= 2) { i0 = sJ0[s]; i1 = sJ1[s]; }
    }
  }
  __syncthreads();

  // phase 3: replay with incoming prefix, park indices in LDS
  int i0 = sI0[t], i1 = sI1[t];
#pragma unroll 4
  for (int i = 0; i < 16; ++i) {
    int l = t * 16 + i;
    if (mask & (1u << i)) { i1 = i0; i0 = l; }
    sIdx0[l] = (short)i0; sIdx1[l] = (short)i1;
  }
  __syncthreads();

  // phase 4: weighted double-gather from V1 (L2-resident) -> out
  const float w1 = w1p[0], w2 = w2p[0];
  const float4* V4 = (const float4*)V1;
  float4* O4 = (float4*)out;
  const int rowQ  = HIDDEN / 4;          // 192 float4 per row
  const int headQ = head * (HEAD_DIM / 4);
  for (int it = 0; it < 256; ++it) {
    int g = it * 256 + t;
    int l = g >> 4;       // row in 0..4095
    int q = g & 15;       // float4 within head slice
    int r0 = sIdx0[l], r1 = sIdx1[l];
    float4 a  = V4[(size_t)(baseM + r0) * rowQ + headQ + q];
    float4 bb = V4[(size_t)(baseM + r1) * rowQ + headQ + q];
    float4 o;
    o.x = w1 * a.x + w2 * bb.x;
    o.y = w1 * a.y + w2 * bb.y;
    o.z = w1 * a.z + w2 * bb.z;
    o.w = w1 * a.w + w2 * bb.w;
    O4[(size_t)(baseM + l) * rowQ + headQ + q] = o;
  }
}

// ---------------------------------------------------------------------------
extern "C" void kernel_launch(void* const* d_in, const int* in_sizes, int n_in,
                              void* d_out, int out_size, void* d_ws, size_t ws_size,
                              hipStream_t stream) {
  const float* H  = (const float*)d_in[0];
  const float* Wk = (const float*)d_in[1];
  const float* bk = (const float*)d_in[2];
  const float* Wv = (const float*)d_in[3];
  const float* bv = (const float*)d_in[4];
  const float* RH = (const float*)d_in[5];
  const float* w1 = (const float*)d_in[6];
  const float* w2 = (const float*)d_in[7];
  float* out = (float*)d_out;

  char* ws = (char*)d_ws;
  const size_t WB  = (size_t)HIDDEN * HIDDEN * sizeof(bf16_t);   // 1,179,648 B
  const size_t V1B = (size_t)M_TOTAL * HIDDEN * sizeof(float);   // 100,663,296 B
  bf16_t* WkB  = (bf16_t*)ws;
  bf16_t* WvB  = (bf16_t*)(ws + WB);
  float*  V1   = (float*)(ws + 2 * WB);
  float*  dotg = (float*)(ws + 2 * WB + V1B);

  wconv_kernel<<<(HIDDEN * HIDDEN) / (256 * 4), 256, 0, stream>>>(Wk, Wv, WkB, WvB);
  kvproj_kernel<<<dim3(M_TOTAL / BM, N_HEADS), 256, 0, stream>>>(
      H, WkB, WvB, bk, bv, RH, V1, dotg);
  scan_gather_kernel<<<dim3(N_HEADS, BSZ), 256, 0, stream>>>(dotg, V1, w1, w2, out);
}
